// MambaBlock_5488968204628
// MI455X (gfx1250) — 8-bit matrix operand order repaired and checked by reading, compiled, not run on hardware
//
#include <hip/hip_runtime.h>
#include <cstdint>

constexpr int Bb = 2, Ll = 2048, Hh = 1024, DI = 2048, Nn = 16, Kc = 4, Rr = 64;
constexpr int TOK = Bb * Ll;

typedef int v8i __attribute__((ext_vector_type(8)));

#if defined(__has_builtin)
#if __has_builtin(__builtin_amdgcn_global_load_async_to_lds_b128)
#define HAVE_ASYNC_LDS 1
#endif
#endif

#ifdef HAVE_ASYNC_LDS
typedef int v4i_b128 __attribute__((vector_size(16)));
typedef __attribute__((address_space(1))) v4i_b128 gv4i_b128;
typedef __attribute__((address_space(3))) v4i_b128 lv4i_b128;
#endif

__device__ inline void wait_asynccnt0() {
#if defined(__has_builtin)
#if __has_builtin(__builtin_amdgcn_s_wait_asynccnt)
  __builtin_amdgcn_s_wait_asynccnt(0);
#else
  asm volatile("s_wait_asynccnt 0" ::: "memory");
#endif
#else
  asm volatile("s_wait_asynccnt 0" ::: "memory");
#endif
}

__device__ inline float block_reduce_sum(float v, float* red) {
  int tid = threadIdx.x;
  red[tid] = v; __syncthreads();
  for (int s = 128; s > 0; s >>= 1) {
    if (tid < s) red[tid] += red[tid + s];
    __syncthreads();
  }
  float r = red[0]; __syncthreads();
  return r;
}
__device__ inline float block_reduce_max(float v, float* red) {
  int tid = threadIdx.x;
  red[tid] = v; __syncthreads();
  for (int s = 128; s > 0; s >>= 1) {
    if (tid < s) red[tid] = fmaxf(red[tid], red[tid + s]);
    __syncthreads();
  }
  float r = red[0]; __syncthreads();
  return r;
}

__device__ inline int8_t quant8(float v, float sc) {
  int q = (int)rintf(v * sc);
  q = q < -128 ? -128 : (q > 127 ? 127 : q);
  return (int8_t)q;
}
__device__ inline float silu(float x) { return x / (1.f + __expf(-x)); }
__device__ inline float softplus(float x) { return x > 20.f ? x : log1pf(__expf(x)); }

constexpr int RED_BLOCKS = 512;

__global__ __launch_bounds__(256) void partial_abs_kernel(const float* __restrict__ w,
                                                          size_t n,
                                                          float* __restrict__ part) {
  __shared__ float red[256];
  float s = 0.f;
  for (size_t i = (size_t)blockIdx.x * 256 + threadIdx.x; i < n;
       i += (size_t)gridDim.x * 256)
    s += fabsf(w[i]);
  float tot = block_reduce_sum(s, red);
  if (threadIdx.x == 0) part[blockIdx.x] = tot;
}

__global__ __launch_bounds__(256) void finalize_meanabs(const float* __restrict__ part,
                                                        int nb, float ninv,
                                                        float* __restrict__ out) {
  __shared__ float red[256];
  float s = 0.f;
  for (int i = threadIdx.x; i < nb; i += 256) s += part[i];
  float tot = block_reduce_sum(s, red);
  if (threadIdx.x == 0) out[0] = fmaxf(tot * ninv, 1e-5f);
}

__global__ __launch_bounds__(256) void quantw_kernel(const float* __restrict__ w,
                                                     const float* __restrict__ scale,
                                                     int8_t* __restrict__ wq, size_t n) {
  size_t i = (size_t)blockIdx.x * 256 + threadIdx.x;
  if (i >= n) return;
  float ws = 1.f / scale[0];
  int q = (int)rintf(w[i] * ws);
  q = q < -1 ? -1 : (q > 1 ? 1 : q);
  wq[i] = (int8_t)q;
}

__global__ __launch_bounds__(256) void rmsnorm_quant(const float* __restrict__ x,
                                                     const float* __restrict__ nw,
                                                     int8_t* __restrict__ xq,
                                                     float* __restrict__ inv_s) {
  __shared__ float red[256];
  const int t = blockIdx.x;
  const float* xr = x + (size_t)t * Hh;
  float ss = 0.f;
  for (int i = threadIdx.x; i < Hh; i += 256) { float v = xr[i]; ss += v * v; }
  float tot = block_reduce_sum(ss, red);
  float rstd = rsqrtf(tot / (float)Hh + 1e-6f);
  float amax = 0.f;
  for (int i = threadIdx.x; i < Hh; i += 256)
    amax = fmaxf(amax, fabsf(xr[i] * rstd * nw[i]));
  amax = block_reduce_max(amax, red);
  float a = fmaxf(amax, 1e-5f);
  float sc = 127.f / a;
  if (threadIdx.x == 0) inv_s[t] = a / 127.f;
  for (int i = threadIdx.x; i < Hh; i += 256)
    xq[(size_t)t * Hh + i] = quant8(xr[i] * rstd * nw[i], sc);
}

template<int KDIM, int NSTRIP>
__global__ __launch_bounds__(256) void gemm_iu8_lds(const int8_t* __restrict__ A,
                                                    const float* __restrict__ inv_s,
                                                    const int8_t* __restrict__ W,
                                                    const float* __restrict__ wscale,
                                                    float* __restrict__ C,
                                                    const float* __restrict__ resid,
                                                    int Nf) {
  __shared__ int8_t Atile[16 * KDIM];
  const int lane = threadIdx.x & 31;
  const int wv   = threadIdx.x >> 5;
  const int m0   = blockIdx.y * 16;
  const int n0   = (blockIdx.x * 8 + wv) * (16 * NSTRIP);
  const int half = lane >> 4;
  const int r16  = lane & 15;

  const int8_t* gsrc = A + (size_t)m0 * KDIM;
#ifdef HAVE_ASYNC_LDS
  for (int i = threadIdx.x; i < KDIM; i += 256) {
    __builtin_amdgcn_global_load_async_to_lds_b128(
        (gv4i_b128*)(void*)(gsrc + (size_t)i * 16),
        (lv4i_b128*)(void*)(Atile + (size_t)i * 16),
        0, 0);
  }
  wait_asynccnt0();
#else
  for (int i = threadIdx.x; i < KDIM; i += 256)
    ((int4*)Atile)[i] = ((const int4*)gsrc)[i];
#endif
  __syncthreads();

  v8i acc[NSTRIP] = {};

#pragma unroll 2
  for (int k0 = 0; k0 < KDIM; k0 += 64) {
    v8i a;
    {
      const int8_t* ap = Atile + r16 * KDIM + half * 16 + k0;
#pragma unroll
      for (int j = 0; j < 4; ++j) {
        int2 t = *(const int2*)(ap + j * 8 + (j >> 1) * 16);
        a[2 * j] = t.x; a[2 * j + 1] = t.y;
      }
    }
#pragma unroll
    for (int i = 0; i < NSTRIP; ++i) {
      const int8_t* bp = W + (size_t)(n0 + i * 16 + r16) * KDIM + k0 + half * 16;
      int4 t0 = *(const int4*)(bp);
      int4 t1 = *(const int4*)(bp + 32);
      v8i bf;
      bf[0] = t0.x; bf[1] = t0.y; bf[2] = t0.z; bf[3] = t0.w;
      bf[4] = t1.x; bf[5] = t1.y; bf[6] = t1.z; bf[7] = t1.w;
      acc[i] = __builtin_amdgcn_wmma_i32_16x16x64_iu8(true, a, true, bf, acc[i],
                                                      false, false);
    }
  }

  const float ws = wscale[0];
#pragma unroll
  for (int i = 0; i < NSTRIP; ++i) {
#pragma unroll
    for (int r = 0; r < 8; ++r) {
      int row = m0 + r + half * 8;
      int col = n0 + i * 16 + r16;
      float v = (float)acc[i][r] * inv_s[row] * ws;
      if (resid) v += resid[(size_t)row * Nf + col];
      C[(size_t)row * Nf + col] = v;
    }
  }
}

template<int NSTRIP>
__global__ __launch_bounds__(256) void gemm_iu8(const int8_t* __restrict__ A,
                                                const float* __restrict__ inv_s,
                                                const int8_t* __restrict__ W,
                                                const float* __restrict__ wscale,
                                                float* __restrict__ C,
                                                const float* __restrict__ resid,
                                                int M, int Nf, int Kdim) {
  const int lane = threadIdx.x & 31;
  const int wave = blockIdx.x * (blockDim.x >> 5) + (threadIdx.x >> 5);
  const int nstrips = Nf / (16 * NSTRIP);
  const int total = (M / 16) * nstrips;
  if (wave >= total) return;
  const int mt = wave / nstrips;
  const int ns = wave % nstrips;
  const int m0 = mt * 16;
  const int n0 = ns * (16 * NSTRIP);
  const int half = lane >> 4;
  const int r16  = lane & 15;

  v8i acc[NSTRIP] = {};

  const int8_t* arow = A + (size_t)(m0 + r16) * Kdim + half * 16;
  for (int k0 = 0; k0 < Kdim; k0 += 64) {
    v8i a;
#pragma unroll
    for (int j = 0; j < 4; ++j) {
      int2 t = *(const int2*)(arow + k0 + j * 8 + (j >> 1) * 16);
      a[2 * j] = t.x; a[2 * j + 1] = t.y;
    }
#pragma unroll
    for (int i = 0; i < NSTRIP; ++i) {
      const int8_t* bp = W + (size_t)(n0 + i * 16 + r16) * Kdim + k0 + half * 16;
      int4 t0 = *(const int4*)(bp);
      int4 t1 = *(const int4*)(bp + 32);
      v8i bf;
      bf[0] = t0.x; bf[1] = t0.y; bf[2] = t0.z; bf[3] = t0.w;
      bf[4] = t1.x; bf[5] = t1.y; bf[6] = t1.z; bf[7] = t1.w;
      acc[i] = __builtin_amdgcn_wmma_i32_16x16x64_iu8(true, a, true, bf, acc[i],
                                                      false, false);
    }
  }

  const float ws = wscale[0];
#pragma unroll
  for (int i = 0; i < NSTRIP; ++i) {
#pragma unroll
    for (int r = 0; r < 8; ++r) {
      int row = m0 + r + half * 8;
      int col = n0 + i * 16 + r16;
      float v = (float)acc[i][r] * inv_s[row] * ws;
      if (resid) v += resid[(size_t)row * Nf + col];
      C[(size_t)row * Nf + col] = v;
    }
  }
}

__global__ __launch_bounds__(256) void conv_silu_quant(const float* __restrict__ xz,
                                                       const float* __restrict__ cw,
                                                       const float* __restrict__ cb,
                                                       float* __restrict__ xc,
                                                       int8_t* __restrict__ xcq,
                                                       float* __restrict__ inv_s) {
  __shared__ float red[256];
  const int t = blockIdx.x;
  const int b = t / Ll, l = t % Ll;
  constexpr int PER = DI / 256;
  float vals[PER];
  float amax = 0.f;
  for (int j = 0; j < PER; ++j) {
    int d = threadIdx.x + j * 256;
    float acc = cb[d];
#pragma unroll
    for (int k = 0; k < Kc; ++k) {
      int ls = l - k;
      if (ls >= 0)
        acc += xz[(size_t)(b * Ll + ls) * (2 * DI) + d] * cw[d * Kc + k];
    }
    float s = silu(acc);
    vals[j] = s;
    xc[(size_t)t * DI + d] = s;
    amax = fmaxf(amax, fabsf(s));
  }
  amax = block_reduce_max(amax, red);
  float a = fmaxf(amax, 1e-5f);
  float sc = 127.f / a;
  if (threadIdx.x == 0) inv_s[t] = a / 127.f;
  for (int j = 0; j < PER; ++j) {
    int d = threadIdx.x + j * 256;
    xcq[(size_t)t * DI + d] = quant8(vals[j], sc);
  }
}

__global__ __launch_bounds__(256) void dt_down_softplus(const float* __restrict__ dtr,
                                                        const float* __restrict__ w,
                                                        const float* __restrict__ bias,
                                                        float* __restrict__ dt) {
  size_t idx = (size_t)blockIdx.x * 256 + threadIdx.x;
  int d = (int)(idx % DI);
  size_t t = idx / DI;
  const float* a  = dtr + t * Rr;
  const float* wr = w + (size_t)d * Rr;
  float acc = bias[d];
#pragma unroll 8
  for (int k = 0; k < Rr; ++k) acc += a[k] * wr[k];
  dt[idx] = softplus(acc);
}

__global__ __launch_bounds__(256) void scan_kernel(const float* __restrict__ dt,
                                                   const float* __restrict__ Bm,
                                                   const float* __restrict__ Cm,
                                                   const float* __restrict__ xc,
                                                   const float* __restrict__ A_log,
                                                   const float* __restrict__ Dp,
                                                   float* __restrict__ ypre) {
  const int b = blockIdx.x / (DI / 256);
  const int d = (blockIdx.x % (DI / 256)) * 256 + threadIdx.x;
  float Av[Nn];
#pragma unroll
  for (int n = 0; n < Nn; ++n) Av[n] = -__expf(A_log[d * Nn + n]);
  float h[Nn] = {};
  const float Dd = Dp[d];
  __shared__ float Bs[Nn], Cs[Nn];
  for (int l = 0; l < Ll; ++l) {
    const int t = b * Ll + l;
    if (threadIdx.x < Nn)            Bs[threadIdx.x]      = Bm[t * Nn + threadIdx.x];
    else if (threadIdx.x < 2 * Nn)   Cs[threadIdx.x - Nn] = Cm[t * Nn + threadIdx.x - Nn];
    __syncthreads();
    const float dtv = dt[(size_t)t * DI + d];
    const float xv  = xc[(size_t)t * DI + d];
    float y = 0.f;
#pragma unroll
    for (int n = 0; n < Nn; ++n) {
      float dA = __expf(Av[n] * dtv);
      h[n] = dA * h[n] + dtv * Bs[n] * xv;
      y += h[n] * Cs[n];
    }
    ypre[(size_t)t * DI + d] = y + xv * Dd;
    __syncthreads();
  }
}

__global__ __launch_bounds__(256) void gate_quant(const float* __restrict__ ypre,
                                                  const float* __restrict__ xz,
                                                  int8_t* __restrict__ yq,
                                                  float* __restrict__ inv_s) {
  __shared__ float red[256];
  const int t = blockIdx.x;
  const float* z = xz + (size_t)t * (2 * DI) + DI;
  constexpr int PER = DI / 256;
  float vals[PER];
  float amax = 0.f;
  for (int j = 0; j < PER; ++j) {
    int d = threadIdx.x + j * 256;
    float g = ypre[(size_t)t * DI + d] * silu(z[d]);
    vals[j] = g;
    amax = fmaxf(amax, fabsf(g));
  }
  amax = block_reduce_max(amax, red);
  float a = fmaxf(amax, 1e-5f);
  float sc = 127.f / a;
  if (threadIdx.x == 0) inv_s[t] = a / 127.f;
  for (int j = 0; j < PER; ++j) {
    int d = threadIdx.x + j * 256;
    yq[(size_t)t * DI + d] = quant8(vals[j], sc);
  }
}

extern "C" void kernel_launch(void* const* d_in, const int* in_sizes, int n_in,
                              void* d_out, int out_size, void* d_ws, size_t ws_size,
                              hipStream_t stream) {
  (void)in_sizes; (void)n_in; (void)out_size; (void)ws_size;
  const float* x         = (const float*)d_in[0];
  const float* norm_w    = (const float*)d_in[1];
  const float* in_proj_w = (const float*)d_in[2];
  const float* conv_w    = (const float*)d_in[3];
  const float* conv_b    = (const float*)d_in[4];
  const float* dt_proj_w = (const float*)d_in[5];
  const float* dt_down_w = (const float*)d_in[6];
  const float* dt_down_b = (const float*)d_in[7];
  const float* B_proj_w  = (const float*)d_in[8];
  const float* C_proj_w  = (const float*)d_in[9];
  const float* A_log     = (const float*)d_in[10];
  const float* Dvec      = (const float*)d_in[11];
  const float* out_proj_w= (const float*)d_in[12];

  char* base = (char*)d_ws;
  size_t off = 0;
  auto carve = [&](size_t bytes) -> char* {
    char* p = base + off;
    off = (off + bytes + 255) & ~(size_t)255;
    return p;
  };
  int8_t* xq_in  = (int8_t*)carve((size_t)TOK * Hh);
  float*  inv_in = (float*) carve((size_t)TOK * 4);
  int8_t* wq_in  = (int8_t*)carve((size_t)2 * DI * Hh);
  int8_t* wq_dt  = (int8_t*)carve((size_t)Rr * DI);
  int8_t* wq_B   = (int8_t*)carve((size_t)Nn * DI);
  int8_t* wq_C   = (int8_t*)carve((size_t)Nn * DI);
  int8_t* wq_out = (int8_t*)carve((size_t)Hh * DI);
  float*  wsc    = (float*) carve(8 * 4);
  float*  redp   = (float*) carve(RED_BLOCKS * 4);
  float*  xz     = (float*) carve((size_t)TOK * 2 * DI * 4);
  float*  xc     = (float*) carve((size_t)TOK * DI * 4);
  int8_t* xcq    = (int8_t*)carve((size_t)TOK * DI);
  float*  inv_xc = (float*) carve((size_t)TOK * 4);
  float*  dtr    = (float*) carve((size_t)TOK * Rr * 4);
  float*  dtb    = (float*) carve((size_t)TOK * DI * 4);
  float*  Bm     = (float*) carve((size_t)TOK * Nn * 4);
  float*  Cm     = (float*) carve((size_t)TOK * Nn * 4);
  float*  ypre   = (float*) carve((size_t)TOK * DI * 4);
  int8_t* yq     = (int8_t*)carve((size_t)TOK * DI);
  float*  inv_y  = (float*) carve((size_t)TOK * 4);

  auto qgrid = [](size_t n) { return (int)((n + 255) / 256); };
  auto ggrid = [](int M, int Nf, int nstrip) {
    int waves = (M / 16) * (Nf / (16 * nstrip));
    return (waves + 7) / 8;
  };
  auto meanabs = [&](const float* w, size_t n, float* out) {
    partial_abs_kernel<<<RED_BLOCKS, 256, 0, stream>>>(w, n, redp);
    finalize_meanabs<<<1, 256, 0, stream>>>(redp, RED_BLOCKS, 1.0f / (float)n, out);
  };

  meanabs(in_proj_w,  (size_t)2 * DI * Hh, wsc + 0);
  quantw_kernel<<<qgrid((size_t)2*DI*Hh), 256, 0, stream>>>(in_proj_w,  wsc + 0, wq_in,  (size_t)2*DI*Hh);
  meanabs(dt_proj_w,  (size_t)Rr * DI,     wsc + 1);
  quantw_kernel<<<qgrid((size_t)Rr*DI),   256, 0, stream>>>(dt_proj_w,  wsc + 1, wq_dt,  (size_t)Rr*DI);
  meanabs(B_proj_w,   (size_t)Nn * DI,     wsc + 2);
  quantw_kernel<<<qgrid((size_t)Nn*DI),   256, 0, stream>>>(B_proj_w,   wsc + 2, wq_B,   (size_t)Nn*DI);
  meanabs(C_proj_w,   (size_t)Nn * DI,     wsc + 3);
  quantw_kernel<<<qgrid((size_t)Nn*DI),   256, 0, stream>>>(C_proj_w,   wsc + 3, wq_C,   (size_t)Nn*DI);
  meanabs(out_proj_w, (size_t)Hh * DI,     wsc + 4);
  quantw_kernel<<<qgrid((size_t)Hh*DI),   256, 0, stream>>>(out_proj_w, wsc + 4, wq_out, (size_t)Hh*DI);

  rmsnorm_quant<<<TOK, 256, 0, stream>>>(x, norm_w, xq_in, inv_in);

  {
    dim3 grid((2 * DI) / (16 * 4) / 8, TOK / 16);
    gemm_iu8_lds<Hh, 4><<<grid, 256, 0, stream>>>(
        xq_in, inv_in, wq_in, wsc + 0, xz, nullptr, 2 * DI);
  }

  conv_silu_quant<<<TOK, 256, 0, stream>>>(xz, conv_w, conv_b, xc, xcq, inv_xc);

  gemm_iu8<4><<<ggrid(TOK, Rr, 4), 256, 0, stream>>>(
      xcq, inv_xc, wq_dt, wsc + 1, dtr, nullptr, TOK, Rr, DI);
  gemm_iu8<1><<<ggrid(TOK, Nn, 1), 256, 0, stream>>>(
      xcq, inv_xc, wq_B, wsc + 2, Bm, nullptr, TOK, Nn, DI);
  gemm_iu8<1><<<ggrid(TOK, Nn, 1), 256, 0, stream>>>(
      xcq, inv_xc, wq_C, wsc + 3, Cm, nullptr, TOK, Nn, DI);

  dt_down_softplus<<<(int)(((size_t)TOK * DI) / 256), 256, 0, stream>>>(
      dtr, dt_down_w, dt_down_b, dtb);

  scan_kernel<<<Bb * (DI / 256), 256, 0, stream>>>(dtb, Bm, Cm, xc, A_log, Dvec, ypre);

  gate_quant<<<TOK, 256, 0, stream>>>(ypre, xz, yq, inv_y);

  {
    dim3 grid(Hh / (16 * 4) / 8, TOK / 16);
    gemm_iu8_lds<DI, 4><<<grid, 256, 0, stream>>>(
        yq, inv_y, wq_out, wsc + 4, (float*)d_out, x, Hh);
  }
}
